// SplineConv_10960756540291
// MI455X (gfx1250) — compile-verified, matrix operand order checked by reading, not run on hardware
//
#include <hip/hip_runtime.h>
#include <math.h>

typedef __attribute__((ext_vector_type(16))) _Float16 v16h;
typedef __attribute__((ext_vector_type(8)))  float    v8f;

constexpr int N_   = 50000;
constexpr int E_   = 800000;
constexpr int FIN_ = 32;
constexpr int FOUT_= 64;
constexpr int K_   = 125;   // kernel points (5^3)
constexpr int S_   = 8;     // 2^3 basis fns per edge
constexpr int KS_  = 5;

// ---------------------------------------------------------------------------
// helpers
// ---------------------------------------------------------------------------
__device__ __forceinline__ void edge_bins(const float* __restrict__ ps,
                                          int wi[8], float bs[8]) {
    float fr[3]; int fl[3];
#pragma unroll
    for (int d = 0; d < 3; ++d) {
        float v = ps[d] * (float)(KS_ - 1);   // kernel_size - is_open
        float f = floorf(v);
        fr[d] = v - f;
        fl[d] = (int)f;
    }
#pragma unroll
    for (int s = 0; s < S_; ++s) {
        float b = 1.0f; int idx = 0; int mul = 1;
#pragma unroll
        for (int d = 0; d < 3; ++d) {
            int bit = (s >> d) & 1;
            b *= bit ? fr[d] : (1.0f - fr[d]);
            int id = fl[d] + bit; if (id >= KS_) id -= KS_;
            idx += id * mul; mul *= KS_;
        }
        wi[s] = idx; bs[s] = b;
    }
}

// ---------------------------------------------------------------------------
// pass 1: zero accumulators
// ---------------------------------------------------------------------------
__global__ void zero_init(float* __restrict__ out, float* __restrict__ deg,
                          unsigned* __restrict__ counts) {
    int i = blockIdx.x * 256 + threadIdx.x;
    if (i < N_ * FOUT_) out[i] = 0.0f;
    if (i < N_)         deg[i] = 0.0f;
    if (i < K_)         counts[i] = 0u;
}

// ---------------------------------------------------------------------------
// pass 2: convert weight fp32 -> f16 pre-swizzled into WMMA B-operand layout.
// B element j of lane L (tile (k,c)): column n = c*16 + (L&15),
// K index kk = (j>>3)*16 + (L>>4)*8 + (j&7).  (ISA 7.12.2, 16-bit operands)
// ---------------------------------------------------------------------------
__global__ void convert_w(const float* __restrict__ w, _Float16* __restrict__ wh) {
    int idx  = blockIdx.x * 256 + threadIdx.x;    // < 126*4*32*16
    int j    = idx & 15;
    int lane = (idx >> 4) & 31;
    int c    = (idx >> 9) & 3;
    int k    = idx >> 11;
    int half = lane >> 4;
    int kk   = (j >> 3) * 16 + half * 8 + (j & 7);
    int n    = c * 16 + (lane & 15);
    wh[idx]  = (_Float16)w[k * (FIN_ * FOUT_) + kk * FOUT_ + n];
}

// ---------------------------------------------------------------------------
// pass 3: per-bin pair counts (LDS-aggregated) + degree
// ---------------------------------------------------------------------------
__global__ void count_edges(const int* __restrict__ ei, const float* __restrict__ ps,
                            float* __restrict__ deg, unsigned* __restrict__ counts) {
    __shared__ unsigned lc[K_];
    int t = threadIdx.x;
    if (t < K_) lc[t] = 0u;
    __syncthreads();
    int e = blockIdx.x * 256 + t;
    if (e < E_) {
        int dst = ei[e];
        atomicAdd(deg + dst, 1.0f);
        int wi[8]; float bs[8];
        edge_bins(ps + e * 3, wi, bs);
#pragma unroll
        for (int s = 0; s < S_; ++s) atomicAdd(&lc[wi[s]], 1u);
    }
    __syncthreads();
    if (t < K_ && lc[t]) atomicAdd(counts + t, lc[t]);
}

// ---------------------------------------------------------------------------
// pass 4: tiny serial scan over 125 bins (rows + 16-row tiles)
// ---------------------------------------------------------------------------
__global__ void scan_bins(const unsigned* __restrict__ counts,
                          unsigned* __restrict__ binStart,
                          unsigned* __restrict__ tileStart,
                          unsigned* __restrict__ cursor,
                          unsigned* __restrict__ total) {
    if (threadIdx.x == 0) {
        unsigned acc = 0, tac = 0;
        for (int k = 0; k < K_; ++k) {
            binStart[k] = acc; cursor[k] = acc; tileStart[k] = tac;
            unsigned cnt = counts[k];
            acc += cnt; tac += (cnt + 15u) >> 4;
        }
        binStart[K_] = acc; tileStart[K_] = tac; *total = tac;
    }
}

// ---------------------------------------------------------------------------
// pass 5: scatter compact records (e*8+s) into bins, two-phase LDS aggregation
// ---------------------------------------------------------------------------
__global__ void scatter_edges(const int* __restrict__ ei, const float* __restrict__ ps,
                              unsigned* __restrict__ cursor, unsigned* __restrict__ recs) {
    __shared__ unsigned lc[K_];
    __shared__ unsigned lb[K_];
    int t = threadIdx.x;
    if (t < K_) lc[t] = 0u;
    __syncthreads();
    int e = blockIdx.x * 256 + t;
    bool act = (e < E_);
    int wi[8]; unsigned ofs[8];
    if (act) {
        float bs[8];
        edge_bins(ps + e * 3, wi, bs);
#pragma unroll
        for (int s = 0; s < S_; ++s) ofs[s] = atomicAdd(&lc[wi[s]], 1u);
    }
    __syncthreads();
    if (t < K_) lb[t] = lc[t] ? atomicAdd(cursor + t, lc[t]) : 0u;
    __syncthreads();
    if (act) {
#pragma unroll
        for (int s = 0; s < S_; ++s)
            recs[lb[wi[s]] + ofs[s]] = (unsigned)(e * 8 + s);
    }
}

// ---------------------------------------------------------------------------
// pass 6: grouped GEMM. One wave per 16-row tile of a bin:
//   A (16x32 f16) = basis-scaled gathered x[src] rows, B = pre-swizzled W[k],
//   4 independent v_wmma_f32_16x16x32_f16 (distinct accumulators -> no false
//   WAW chain), then 32 straight-line global f32 atomics into out[dst].
// Invalid tail rows carry basis==0 -> exact-zero C rows -> harmless +0.0 adds.
// ---------------------------------------------------------------------------
__global__ __launch_bounds__(128) void edge_gemm(
        const float* __restrict__ x, const int* __restrict__ ei,
        const float* __restrict__ ps, const unsigned* __restrict__ recs,
        const unsigned* __restrict__ binStart, const unsigned* __restrict__ tileStart,
        const unsigned* __restrict__ total, const _Float16* __restrict__ wh,
        float* __restrict__ out) {
    int wid = threadIdx.x >> 5;
    int L   = threadIdx.x & 31;
    unsigned g = blockIdx.x * 4u + (unsigned)wid;
    if (g >= *total) return;                          // no block barriers below

    // binary search: largest k with tileStart[k] <= g (tileStart[125]=total>g)
    unsigned lo = 0, hi = K_;
    while (hi - lo > 1u) {
        unsigned mid = (lo + hi) >> 1;
        if (tileStart[mid] <= g) lo = mid; else hi = mid;
    }
    unsigned k       = lo;
    unsigned rowBase = binStart[k] + 16u * (g - tileStart[k]);
    unsigned binEnd  = binStart[k + 1];

    int M = L & 15, half = L >> 4;

    // --- kick off B-operand loads early: 4 independent 32x16 tiles of W[k]
    const v16h* wv = (const v16h*)wh;
    v16h b0 = wv[(k * 4u + 0u) * 32u + (unsigned)L];
    v16h b1 = wv[(k * 4u + 1u) * 32u + (unsigned)L];
    v16h b2 = wv[(k * 4u + 2u) * 32u + (unsigned)L];
    v16h b3 = wv[(k * 4u + 3u) * 32u + (unsigned)L];

    // --- A operand: lane L carries row M, K-slices {half*8..+7, 16+half*8..+7}
    unsigned rowIdx = rowBase + (unsigned)M;
    bool     valid  = rowIdx < binEnd;
    unsigned rec    = valid ? recs[rowIdx] : 0u;
    unsigned e      = rec >> 3;
    int      s      = (int)(rec & 7u);
    int      src    = valid ? ei[E_ + e] : 0;
    int      dstM   = valid ? ei[e]      : 0;   // dst of row M (this lane)
    float    basis  = 0.0f;
    if (valid) {
        float b = 1.0f;
#pragma unroll
        for (int d = 0; d < 3; ++d) {
            float v  = ps[e * 3 + d] * (float)(KS_ - 1);
            float fr = v - floorf(v);
            b *= ((s >> d) & 1) ? fr : (1.0f - fr);
        }
        basis = b;
    }
    const float4* xp = (const float4*)(x + (size_t)src * FIN_);
    float4 q0 = xp[half * 2 + 0], q1 = xp[half * 2 + 1];
    float4 q2 = xp[half * 2 + 4], q3 = xp[half * 2 + 5];
    float fa[16] = { q0.x,q0.y,q0.z,q0.w, q1.x,q1.y,q1.z,q1.w,
                     q2.x,q2.y,q2.z,q2.w, q3.x,q3.y,q3.z,q3.w };
    v16h a;
#pragma unroll
    for (int i = 0; i < 16; ++i) a[i] = (_Float16)(fa[i] * basis);

    // --- dst per output row via wave32 lane-permute (C VGPR r = row r+8*half;
    //     that row's dst lives in lane (r+8*half) as dstM)
    unsigned rofs[8];
#pragma unroll
    for (int r = 0; r < 8; ++r) {
        int d2 = __shfl(dstM, r + 8 * half, 32);
        rofs[r] = (unsigned)d2 * (unsigned)FOUT_ + (unsigned)M;
    }

    // --- 4 independent WMMAs (one 16x64 output tile)
    v8f z = {0.f,0.f,0.f,0.f,0.f,0.f,0.f,0.f};
    v8f acc0 = __builtin_amdgcn_wmma_f32_16x16x32_f16(false, a, false, b0, (short)0, z, false, false);
    v8f acc1 = __builtin_amdgcn_wmma_f32_16x16x32_f16(false, a, false, b1, (short)0, z, false, false);
    v8f acc2 = __builtin_amdgcn_wmma_f32_16x16x32_f16(false, a, false, b2, (short)0, z, false, false);
    v8f acc3 = __builtin_amdgcn_wmma_f32_16x16x32_f16(false, a, false, b3, (short)0, z, false, false);

#pragma unroll
    for (int r = 0; r < 8; ++r) atomicAdd(out + rofs[r] +  0u, acc0[r]);
#pragma unroll
    for (int r = 0; r < 8; ++r) atomicAdd(out + rofs[r] + 16u, acc1[r]);
#pragma unroll
    for (int r = 0; r < 8; ++r) atomicAdd(out + rofs[r] + 32u, acc2[r]);
#pragma unroll
    for (int r = 0; r < 8; ++r) atomicAdd(out + rofs[r] + 48u, acc3[r]);
}

// ---------------------------------------------------------------------------
// pass 7: mean-divide + root WMMA (W[125]) + bias.  Block = 16-row tile,
// wave wid handles column tile wid. Each (n,o) touched exactly once.
// ---------------------------------------------------------------------------
__global__ __launch_bounds__(128) void root_fin(
        const float* __restrict__ x, const float* __restrict__ deg,
        const _Float16* __restrict__ wh, const float* __restrict__ bias,
        float* __restrict__ out) {
    int wid = threadIdx.x >> 5;
    int L   = threadIdx.x & 31;
    int M = L & 15, half = L >> 4;
    int row = blockIdx.x * 16 + M;

    const float4* xp = (const float4*)(x + (size_t)row * FIN_);
    float4 q0 = xp[half * 2 + 0], q1 = xp[half * 2 + 1];
    float4 q2 = xp[half * 2 + 4], q3 = xp[half * 2 + 5];
    float fa[16] = { q0.x,q0.y,q0.z,q0.w, q1.x,q1.y,q1.z,q1.w,
                     q2.x,q2.y,q2.z,q2.w, q3.x,q3.y,q3.z,q3.w };
    v16h a;
#pragma unroll
    for (int i = 0; i < 16; ++i) a[i] = (_Float16)fa[i];

    const v16h* wv = (const v16h*)wh;
    v16h b = wv[((unsigned)K_ * 4u + (unsigned)wid) * 32u + (unsigned)L];
    v8f acc = {0.f,0.f,0.f,0.f,0.f,0.f,0.f,0.f};
    acc = __builtin_amdgcn_wmma_f32_16x16x32_f16(
              false, a, false, b, (short)0, acc, false, false);

    int   col = wid * 16 + M;
    float bc  = bias[col];
#pragma unroll
    for (int r = 0; r < 8; ++r) {
        int n = blockIdx.x * 16 + r + 8 * half;
        float inv = 1.0f / fmaxf(deg[n], 1.0f);
        float* op = out + (size_t)n * FOUT_ + col;
        *op = (*op) * inv + acc[r] + bc;
    }
}

// ---------------------------------------------------------------------------
extern "C" void kernel_launch(void* const* d_in, const int* in_sizes, int n_in,
                              void* d_out, int out_size, void* d_ws, size_t ws_size,
                              hipStream_t stream) {
    const float* x      = (const float*)d_in[0];
    const int*   ei     = (const int*)d_in[1];     // [2,E]: row0=dst, row1=src
    const float* pseudo = (const float*)d_in[2];
    const float* weight = (const float*)d_in[3];   // [126,32,64]
    const float* bias   = (const float*)d_in[4];
    float*       out    = (float*)d_out;           // [N,64] fp32

    // workspace carve-out (~26.4 MB total), 256B-aligned slabs
    char*  wsb = (char*)d_ws;
    size_t off = 0;
    auto take = [&](size_t bytes) -> void* {
        void* p = wsb + off;
        off += (bytes + 255u) & ~(size_t)255u;
        return p;
    };
    _Float16* wh        = (_Float16*)take((size_t)(K_ + 1) * 4 * 32 * 16 * 2); // 516 KB
    float*    deg       = (float*)   take((size_t)N_ * 4);
    unsigned* counts    = (unsigned*)take(128 * 4);
    unsigned* binStart  = (unsigned*)take(128 * 4);
    unsigned* tileStart = (unsigned*)take(128 * 4);
    unsigned* cursor    = (unsigned*)take(128 * 4);
    unsigned* total     = (unsigned*)take(256);
    unsigned* recs      = (unsigned*)take((size_t)E_ * S_ * 4);                // 25.6 MB
    (void)off; (void)ws_size; (void)in_sizes; (void)n_in; (void)out_size;

    zero_init   <<<(N_ * FOUT_ + 255) / 256, 256, 0, stream>>>(out, deg, counts);
    convert_w   <<<((K_ + 1) * 4 * 32 * 16) / 256, 256, 0, stream>>>(weight, wh);
    count_edges <<<(E_ + 255) / 256, 256, 0, stream>>>(ei, pseudo, deg, counts);
    scan_bins   <<<1, 32, 0, stream>>>(counts, binStart, tileStart, cursor, total);
    scatter_edges<<<(E_ + 255) / 256, 256, 0, stream>>>(ei, pseudo, cursor, recs);

    const int maxTiles = (E_ * S_) / 16 + K_;      // 400125 upper bound
    edge_gemm   <<<(maxTiles + 3) / 4, 128, 0, stream>>>(
        x, ei, pseudo, recs, binStart, tileStart, total, wh, out);
    root_fin    <<<N_ / 16, 128, 0, stream>>>(x, deg, wh, bias, out);
}
